// GCN_31129922962006
// MI455X (gfx1250) — matrix operand order repaired and checked by reading, not compiled, not run on hardware
//
#include <hip/hip_runtime.h>
#include <hip/hip_bf16.h>

// ---------------------------------------------------------------------------
// GCN forward on MI455X (gfx1250, wave32):
//   H1 = A @ X            (COO spmm, native global_atomic_add_f32 scatter)
//   H2 = relu(H1@W1 + b1) (v_wmma_f32_16x16x32_bf16, f32 accumulate)
//   H3 = A @ H2           (spmm, H3 aliases H1)
//   out = H3@W2 + b2      (WMMA)
//
// Roofline: FLOPs < 3 GFLOP (negligible). Cost is SpMM gather (800K edges x
// 512 B) + 102M fp32 atomic adds, all hitting the 192 MB L2 (operand matrices
// are 25.6 MB). Native non-returning atomics are mandatory; WMMA precision
// choice (bf16 in, f32 acc) is free perf-wise and uses the fast matrix path.
// ---------------------------------------------------------------------------

typedef __attribute__((ext_vector_type(16))) __bf16 v16bf;
typedef __attribute__((ext_vector_type(8)))  float  v8f;

// Guaranteed single non-returning fp32 atomic add (STOREcnt-tracked).
__device__ __forceinline__ void atomic_fadd_f32(float* p, float v) {
    asm volatile("global_atomic_add_f32 %0, %1, off"
                 :: "v"(p), "v"(v)
                 : "memory");
}

// ---------------------------------------------------------------------------
// Pack a row-major f32 weight matrix W[K=128][ncols] into bf16 B-fragments in
// the VGPR layout of V_WMMA_F32_16X16X32_BF16's B operand:
//   frag index = ((nt*4 + kt)*32 + lane)*16 + i
//   lane n<16  holds column nt*16+n, K = kt*32 + (i&7) + 16*(i>>3)      (i = 0..15)
//   lane n+16  holds column nt*16+n, K = kt*32 + 8 + (i&7) + 16*(i>>3)
// ---------------------------------------------------------------------------
__global__ void gcn_pack_w(const float* __restrict__ W, __bf16* __restrict__ Wf,
                           int ncols, int total) {
    int idx = blockIdx.x * blockDim.x + threadIdx.x;
    if (idx >= total) return;
    int i    = idx & 15;
    int lane = (idx >> 4) & 31;
    int kt   = (idx >> 9) & 3;
    int nt   = idx >> 11;
    int g = lane >> 4;
    int n = lane & 15;
    int k   = kt * 32 + g * 8 + (i & 7) + ((i >> 3) << 4);
    int col = nt * 16 + n;
    Wf[idx] = (__bf16)W[k * ncols + col];
}

// ---------------------------------------------------------------------------
// COO SpMM: one wave per edge. Lane l gathers features [4l,4l+4) of X[col]
// as one float4 (coalesced 512 B per edge, L2-resident) and scatter-adds into
// Y[row] with 4 native fp32 global atomics.
// ---------------------------------------------------------------------------
__global__ void gcn_spmm(const int* __restrict__ erow, const int* __restrict__ ecol,
                         const float* __restrict__ eval, const float* __restrict__ Xin,
                         float* __restrict__ Y, int E) {
    int wave = blockIdx.x * (blockDim.x >> 5) + (threadIdx.x >> 5);
    int lane = threadIdx.x & 31;
    if (wave >= E) return;
    int r = __builtin_amdgcn_readfirstlane(erow[wave]);
    int c = __builtin_amdgcn_readfirstlane(ecol[wave]);
    float v = eval[wave];
    const float4* xr = (const float4*)(Xin + (size_t)c * 128);
    float4 x = xr[lane];
    float* y = Y + (size_t)r * 128 + lane * 4;
    atomic_fadd_f32(y + 0, v * x.x);
    atomic_fadd_f32(y + 1, v * x.y);
    atomic_fadd_f32(y + 2, v * x.z);
    atomic_fadd_f32(y + 3, v * x.w);
}

// ---------------------------------------------------------------------------
// Dense GEMM out[M x ncols] = act(A[M x 128] @ W + bias).
// Block = 256 threads = 8 waves; each wave owns a 16-row strip and computes
// all NT 16-col tiles, K = 128 in 4 K-tiles of 32.
// A fragments (16x32 bf16, ISA A-layout: lane m=l&15, lane group g picks the
// K sub-blocks g*8..g*8+7 and 16+g*8..16+g*8+7 of each K-tile) are converted
// f32->bf16 in registers (v_cvt_pk_bf16_f32) and reused across all NT tiles.
// ---------------------------------------------------------------------------
template <int NT, bool RELU>
__global__ void gcn_gemm(const float* __restrict__ A, const __bf16* __restrict__ Wf,
                         const float* __restrict__ bias, float* __restrict__ out,
                         int M, int ncols) {
    int wave = threadIdx.x >> 5;
    int lane = threadIdx.x & 31;
    int row_base = (blockIdx.x * 8 + wave) * 16;
    if (row_base >= M) return;
    int g = lane >> 4;
    int n = lane & 15;

    // --- load + convert A fragments for the 4 K-tiles -----------------------
    v16bf afrag[4];
    const float* arow = A + (size_t)(row_base + n) * 128;
#pragma unroll
    for (int kt = 0; kt < 4; ++kt) {
        const float* p = arow + kt * 32 + g * 8;
#pragma unroll
        for (int h = 0; h < 2; ++h) {          // h=0: K+0..7, h=1: K+16..23
            float4 q0 = *(const float4*)(p + h * 16);
            float4 q1 = *(const float4*)(p + h * 16 + 4);
            afrag[kt][h * 8 + 0] = (__bf16)q0.x;
            afrag[kt][h * 8 + 1] = (__bf16)q0.y;
            afrag[kt][h * 8 + 2] = (__bf16)q0.z;
            afrag[kt][h * 8 + 3] = (__bf16)q0.w;
            afrag[kt][h * 8 + 4] = (__bf16)q1.x;
            afrag[kt][h * 8 + 5] = (__bf16)q1.y;
            afrag[kt][h * 8 + 6] = (__bf16)q1.z;
            afrag[kt][h * 8 + 7] = (__bf16)q1.w;
        }
    }

    // --- accumulate: NT n-tiles x 4 k-tiles of v_wmma_f32_16x16x32_bf16 -----
    const v16bf* Wv = (const v16bf*)Wf;
    v8f acc[NT] = {};
#pragma unroll
    for (int nt = 0; nt < NT; ++nt) {
#pragma unroll
        for (int kt = 0; kt < 4; ++kt) {
            v16bf bfrag = Wv[(nt * 4 + kt) * 32 + lane];
            acc[nt] = __builtin_amdgcn_wmma_f32_16x16x32_bf16(
                false, afrag[kt], false, bfrag, (short)0, acc[nt], false, false);
        }
    }

    // --- epilogue: bias (+ReLU) and store (C/D layout: vgpr i -> M=i+8g) ----
#pragma unroll
    for (int nt = 0; nt < NT; ++nt) {
        int colb = nt * 16 + n;
        float bv = bias[colb];
#pragma unroll
        for (int i = 0; i < 8; ++i) {
            int row = row_base + i + 8 * g;
            float vv = acc[nt][i] + bv;
            if (RELU) vv = fmaxf(vv, 0.0f);
            out[(size_t)row * ncols + colb] = vv;
        }
    }
}

// ---------------------------------------------------------------------------
extern "C" void kernel_launch(void* const* d_in, const int* in_sizes, int n_in,
                              void* d_out, int out_size, void* d_ws, size_t ws_size,
                              hipStream_t stream) {
    const float* X  = (const float*)d_in[0];
    const int* erow = (const int*)d_in[1];
    const int* ecol = (const int*)d_in[2];
    const float* ev = (const float*)d_in[3];
    const float* W1 = (const float*)d_in[4];
    const float* b1 = (const float*)d_in[5];
    const float* W2 = (const float*)d_in[6];
    const float* b2 = (const float*)d_in[7];

    const int M = in_sizes[0] / 128;  // 50000 nodes
    const int E = in_sizes[1];        // 800000 edges

    // workspace: H1 (doubles as H3) | H2 | packed bf16 weights
    float*  H1  = (float*)d_ws;
    float*  H2  = H1 + (size_t)M * 128;
    __bf16* W1f = (__bf16*)(H2 + (size_t)M * 128);
    __bf16* W2f = W1f + 8 * 4 * 32 * 16;

    const int w1tot = 8 * 4 * 32 * 16;  // 8 n-tiles (N=128)
    const int w2tot = 4 * 4 * 32 * 16;  // 4 n-tiles (N=64)
    gcn_pack_w<<<(w1tot + 255) / 256, 256, 0, stream>>>(W1, W1f, 128, w1tot);
    gcn_pack_w<<<(w2tot + 255) / 256, 256, 0, stream>>>(W2, W2f, 64, w2tot);

    // H1 = A @ X
    hipMemsetAsync(H1, 0, (size_t)M * 128 * sizeof(float), stream);
    gcn_spmm<<<(E + 7) / 8, 256, 0, stream>>>(erow, ecol, ev, X, H1, E);

    // H2 = relu(H1 @ W1 + b1)
    gcn_gemm<8, true><<<(M + 127) / 128, 256, 0, stream>>>(H1, W1f, b1, H2, M, 128);

    // H3 (= H1 buffer) = A @ H2
    hipMemsetAsync(H1, 0, (size_t)M * 128 * sizeof(float), stream);
    gcn_spmm<<<(E + 7) / 8, 256, 0, stream>>>(erow, ecol, ev, H2, H1, E);

    // out = H3 @ W2 + b2
    gcn_gemm<4, false><<<(M + 127) / 128, 256, 0, stream>>>(H1, W2f, b2,
                                                            (float*)d_out, M, 64);
}
